// LinkMLPPredictor_17540646437589
// MI455X (gfx1250) — matrix operand order repaired and checked by reading, not compiled, not run on hardware
//
#include <hip/hip_runtime.h>

typedef __attribute__((ext_vector_type(16))) __bf16 v16bf;
typedef __attribute__((ext_vector_type(8)))  float  v8f;
typedef __attribute__((ext_vector_type(4)))  float  v4f;

#define IN_DIM 128
#define HID    128

// f32 -> bf16 native conversion (lowers to v_cvt_pk_bf16_f32 pairs)
static __device__ __forceinline__ __bf16 f2bf(float f) {
    return static_cast<__bf16>(f);
}

__global__ __launch_bounds__(256) void link_mlp_wmma_kernel(
    const float* __restrict__ pat,      // [N_NODES, 128]
    const float* __restrict__ cond,     // [N_NODES, 128]
    const long long* __restrict__ eidx, // [2, E]
    const float* __restrict__ W1,       // [256, 128] row-major (k-major)
    const float* __restrict__ b1,       // [128]
    const float* __restrict__ W2,       // [128]
    const float* __restrict__ b2,       // [1]
    float* __restrict__ out,            // [E]
    int E)
{
    // W1 transposed to bf16 in LDS: sWt[n][k], n in [0,128), k in [0,256)
    __shared__ __align__(32) __bf16 sWt[HID * 2 * IN_DIM]; // 64 KB
    __shared__ float sB1[HID];
    __shared__ float sW2[HID];

    const int tid = threadIdx.x;
    for (int i = tid; i < HID * 2 * IN_DIM; i += 256) {
        int n = i & (HID - 1);
        int k = i >> 7;
        sWt[n * (2 * IN_DIM) + k] = f2bf(W1[k * HID + n]); // coalesced global read
    }
    if (tid < HID) {
        sB1[tid] = b1[tid];
        sW2[tid] = W2[tid];
    }
    __syncthreads();

    const int lane = tid & 31;
    const int wave = tid >> 5;
    const int m16  = lane & 15;   // A row / B,C column within tile
    const int hi   = lane >> 4;   // lane half
    const int nwaves = gridDim.x * 8;
    const int tiles  = (E + 31) >> 5;   // 32 edges per wave-iteration
    const float bias2 = b2[0];

    for (int tile = blockIdx.x * 8 + wave; tile < tiles; tile += nwaves) {
        const int ebase = tile * 32;

        // two 16-edge blocks (u = 0, 1)
        int e0 = ebase + m16;        if (e0 >= E) e0 = E - 1;
        int e1 = ebase + 16 + m16;   if (e1 >= E) e1 = E - 1;
        const float* srow0 = pat  + eidx[e0] * IN_DIM;
        const float* drow0 = cond + eidx[(long long)E + e0] * IN_DIM;
        const float* srow1 = pat  + eidx[e1] * IN_DIM;
        const float* drow1 = cond + eidx[(long long)E + e1] * IN_DIM;

        // ---- A fragments: a[u*8 + (0..3)] = src K-steps, a[u*8 + (4..7)] = dst ----
        // Contiguous k order used here (same as B, not the ISA table order): lane<16 holds K kb+0..15; lane>=16 holds K kb+16..31
        v16bf a[16];
        #pragma unroll
        for (int u = 0; u < 2; u++) {
            const float* sr = u ? srow1 : srow0;
            const float* dr = u ? drow1 : drow0;
            #pragma unroll
            for (int ks = 0; ks < 4; ks++) {
                const int o1 = ks * 32 + hi * 16;
                const int o2 = o1 + 8;
                v4f s0 = *(const v4f*)(sr + o1);
                v4f s1 = *(const v4f*)(sr + o1 + 4);
                v4f s2 = *(const v4f*)(sr + o2);
                v4f s3 = *(const v4f*)(sr + o2 + 4);
                v4f d0 = *(const v4f*)(dr + o1);
                v4f d1 = *(const v4f*)(dr + o1 + 4);
                v4f d2 = *(const v4f*)(dr + o2);
                v4f d3 = *(const v4f*)(dr + o2 + 4);
                v16bf av, dv;
                #pragma unroll
                for (int j = 0; j < 4; j++) {
                    av[j]      = f2bf(s0[j]);  av[4 + j]  = f2bf(s1[j]);
                    av[8 + j]  = f2bf(s2[j]);  av[12 + j] = f2bf(s3[j]);
                    dv[j]      = f2bf(d0[j]);  dv[4 + j]  = f2bf(d1[j]);
                    dv[8 + j]  = f2bf(d2[j]);  dv[12 + j] = f2bf(d3[j]);
                }
                a[u * 8 + ks]     = av;
                a[u * 8 + 4 + ks] = dv;
            }
        }

        float logitp0[8], logitp1[8];
        #pragma unroll
        for (int mi = 0; mi < 8; mi++) { logitp0[mi] = 0.f; logitp1[mi] = 0.f; }

        // ---- 8 output N-tiles; each B fragment feeds TWO independent WMMA chains ----
        #pragma unroll 1
        for (int t = 0; t < 8; t++) {
            const int n = t * 16 + m16;
            const float b1v = sB1[n];
            v8f acc0, acc1;
            #pragma unroll
            for (int mi = 0; mi < 8; mi++) { acc0[mi] = b1v; acc1[mi] = b1v; }

            // B layout: lane<16 holds K kb+0..15, lane>=16 holds K kb+16..31.
            // dst half of W1 starts at k=128 = 4*32, so kb = j*32 + hi*16 for j=0..7.
            const __bf16* wrow = &sWt[n * (2 * IN_DIM)];
            #pragma unroll
            for (int j = 0; j < 8; j++) {
                const v16bf bf = *(const v16bf*)(wrow + j * 32 + hi * 16);
                acc0 = __builtin_amdgcn_wmma_f32_16x16x32_bf16(
                           false, a[j],     false, bf, (short)0, acc0, false, false);
                acc1 = __builtin_amdgcn_wmma_f32_16x16x32_bf16(
                           false, a[8 + j], false, bf, (short)0, acc1, false, false);
            }

            // ReLU + second-layer dot folded per lane
            const float w2v = sW2[n];
            #pragma unroll
            for (int mi = 0; mi < 8; mi++) {
                float h0 = acc0[mi]; h0 = h0 > 0.f ? h0 : 0.f;
                float h1 = acc1[mi]; h1 = h1 > 0.f ? h1 : 0.f;
                logitp0[mi] = fmaf(h0, w2v, logitp0[mi]);
                logitp1[mi] = fmaf(h1, w2v, logitp1[mi]);
            }
        }

        // ---- reduce over the 16 columns held across lanes within each half ----
        #pragma unroll
        for (int mi = 0; mi < 8; mi++) {
            float v0 = logitp0[mi];
            float v1 = logitp1[mi];
            v0 += __shfl_xor(v0, 1, 32);
            v1 += __shfl_xor(v1, 1, 32);
            v0 += __shfl_xor(v0, 2, 32);
            v1 += __shfl_xor(v1, 2, 32);
            v0 += __shfl_xor(v0, 4, 32);
            v1 += __shfl_xor(v1, 4, 32);
            v0 += __shfl_xor(v0, 8, 32);
            v1 += __shfl_xor(v1, 8, 32);
            logitp0[mi] = v0;
            logitp1[mi] = v1;
        }

        if (m16 == 0) { // lane 0 -> rows 0..7, lane 16 -> rows 8..15 (per block)
            #pragma unroll
            for (int mi = 0; mi < 8; mi++) {
                const int eo0 = ebase + hi * 8 + mi;
                const int eo1 = ebase + 16 + hi * 8 + mi;
                if (eo0 < E) {
                    const float x = logitp0[mi] + bias2;
                    out[eo0] = 1.f / (1.f + __expf(-x));
                }
                if (eo1 < E) {
                    const float x = logitp1[mi] + bias2;
                    out[eo1] = 1.f / (1.f + __expf(-x));
                }
            }
        }
    }
}

extern "C" void kernel_launch(void* const* d_in, const int* in_sizes, int n_in,
                              void* d_out, int out_size, void* d_ws, size_t ws_size,
                              hipStream_t stream) {
    const float*     pat  = (const float*)d_in[0];
    const float*     cond = (const float*)d_in[1];
    const long long* eidx = (const long long*)d_in[2];
    const float*     W1   = (const float*)d_in[3];
    const float*     b1   = (const float*)d_in[4];
    const float*     W2   = (const float*)d_in[5];
    const float*     b2   = (const float*)d_in[6];
    float*           out  = (float*)d_out;

    const int E = in_sizes[2] / 2;
    const int tiles = (E + 31) / 32;
    int blocks = (tiles + 7) / 8;
    if (blocks > 1024) blocks = 1024;

    link_mlp_wmma_kernel<<<blocks, 256, 0, stream>>>(pat, cond, eidx, W1, b1, W2, b2, out, E);
}
